// EMAVectorQuantizer_40931038330996
// MI455X (gfx1250) — matrix operand order repaired and checked by reading, not compiled, not run on hardware
//
#include <hip/hip_runtime.h>
#include <hip/hip_bf16.h>
#include <stdint.h>

typedef __attribute__((ext_vector_type(16))) _Float16 v16h;
typedef __attribute__((ext_vector_type(8)))  float    v8f;
typedef __attribute__((ext_vector_type(4)))  int      v4i32;

#define D_DIM   256
#define K_CODES 2048
#define N_ROWS  16384
#define ZS      258       // f16 LDS row stride (halves): 129 dwords -> +1 bank/row, conflict-free
#define QS      257       // f32 LDS row stride (floats): conflict-free
#define SLOT_DW (8 * 264) // one swizzled 16-code tile: 8 kb-blocks x (32 lanes x 8 dw + 8 pad)

union Frag16 { v16h h; uint32_t u[8]; };

__device__ __forceinline__ uint32_t pack2h(float a, float b) {
  union { _Float16 h[2]; uint32_t u; } p;
  p.h[0] = (_Float16)a; p.h[1] = (_Float16)b;
  return p.u;
}

// LDS byte offset of a __shared__ object (generic -> AS3 -> int).
#define LDS_OFF(p) ((uint32_t)(uintptr_t)(__attribute__((address_space(3))) const void*)(p))

// Async global->LDS 16B copy: each lane picks its own global source and LDS dest.
__device__ __forceinline__ void async_b128(const uint32_t* gsrc, uint32_t* ldst) {
#if __has_builtin(__builtin_amdgcn_global_load_async_to_lds_b128)
  __builtin_amdgcn_global_load_async_to_lds_b128(
      (__attribute__((address_space(1))) v4i32*)gsrc,
      (__attribute__((address_space(3))) v4i32*)ldst,
      0, 0);
#else
  asm volatile("global_load_async_to_lds_b128 %0, %1, off"
               :: "v"(LDS_OFF(ldst)), "v"(gsrc) : "memory");
#endif
}

__device__ __forceinline__ void wait_asynccnt8() {
#if __has_builtin(__builtin_amdgcn_s_wait_asynccnt)
  __builtin_amdgcn_s_wait_asynccnt(8);
#else
  asm volatile("s_wait_asynccnt 8" ::: "memory");
#endif
}

// ------------------------------------------------------------------
// Kernel 0: codebook f32 -> packed f16 rows + ||c||^2 per code.
// ------------------------------------------------------------------
__global__ void __launch_bounds__(256) vq_prep(const float* __restrict__ cb,
                                               uint32_t* __restrict__ cb16,
                                               float* __restrict__ cnorm) {
  const int gid  = blockIdx.x * 256 + threadIdx.x;
  const int k    = gid >> 5;     // one wave per codebook row
  const int lane = gid & 31;

  const float4* row = (const float4*)(cb + (size_t)k * D_DIM);
  float4 e0 = row[lane];
  float4 e1 = row[lane + 32];

  uint32_t* orow = cb16 + (size_t)k * (D_DIM / 2);
  orow[lane * 2 + 0]      = pack2h(e0.x, e0.y);
  orow[lane * 2 + 1]      = pack2h(e0.z, e0.w);
  orow[64 + lane * 2 + 0] = pack2h(e1.x, e1.y);
  orow[64 + lane * 2 + 1] = pack2h(e1.z, e1.w);

  float ss = e0.x * e0.x + e0.y * e0.y + e0.z * e0.z + e0.w * e0.w
           + e1.x * e1.x + e1.y * e1.y + e1.z * e1.z + e1.w * e1.w;
  #pragma unroll
  for (int off = 1; off < 32; off <<= 1) ss += __shfl_xor(ss, off, 32);
  if (lane == 0) cnorm[k] = ss;
}

// ------------------------------------------------------------------
// Kernel 1: fused distance GEMM (WMMA f16->f32) + running argmin.
// 256 blocks x 128 threads (4 waves). Each wave owns 16 z rows.
// Codebook tiles arrive via async global->LDS DMA, pre-swizzled into
// B-fragment order so each lane's fragment is 2x ds_load_b128.
// ------------------------------------------------------------------
__global__ void __launch_bounds__(128) vq_argmin(const float* __restrict__ z,
                                                 const uint32_t* __restrict__ cb16,
                                                 const float* __restrict__ cnorm,
                                                 int* __restrict__ idx_out) {
  __shared__ _Float16 zsh[64 * ZS];        // 33,024 B : 64-row z strip, f16
  __shared__ uint32_t csh[3][SLOT_DW];     // 25,344 B : async tile ring, swizzled

  const int t    = threadIdx.x;
  const int row0 = blockIdx.x * 64;
  const int b    = row0 >> 10;             // strip lies inside one batch image
  const int hw0  = row0 & 1023;
  const float* zg = z + (size_t)b * (D_DIM * 1024) + hw0;

  // Stage z strip: z[b, d, hw0+r] -> zsh[r][d] (f16). Coalesced global reads.
  {
    const int r = t & 63;
    for (int d = (t >> 6); d < D_DIM; d += 2) {
      float v = zg[(size_t)d * 1024 + r];
      zsh[r * ZS + d] = (_Float16)v;
    }
  }

  const int lane   = t & 31;
  const int wave   = t >> 5;
  const int laneHi = lane >> 4;
  const int nl     = lane & 15;

  // Async chunk mapping: chunk c = p*128 + t covers global dwords tile_base + 4c..4c+3
  // (contiguous => each async b128 instruction reads 512 contiguous bytes).
  // Consumer coords of that chunk:  n = c>>5, kb = (c&31)>>2, laneHi = (c>>1)&1, half = c&1.
  // LDS dest (fragment-major, 4-dword skew per 8-lane group for bank spread):
  //   dw = kb*264 + clane*8 + ((clane>>3)&1)*4 + half*4,  clane = n + 16*laneHi.
  int ldw[4];
  #pragma unroll
  for (int p = 0; p < 4; ++p) {
    int c     = p * 128 + t;
    int n     = c >> 5;
    int kb    = (c & 31) >> 2;
    int chi   = (c >> 1) & 1;
    int half  = c & 1;
    int clane = n + (chi << 4);
    ldw[p] = kb * 264 + clane * 8 + ((clane >> 3) & 1) * 4 + half * 4;
  }
  const uint32_t* gthread = cb16 + t * 4;

  // Prologue: issue tiles 0 and 1 into slots 0 and 1.
  #pragma unroll
  for (int p = 0; p < 4; ++p) async_b128(gthread + p * 512, &csh[0][ldw[p]]);
  #pragma unroll
  for (int p = 0; p < 4; ++p) async_b128(gthread + 2048 + p * 512, &csh[1][ldw[p]]);

  __syncthreads();   // zsh ready

  // Per-wave A fragments (rows wave*16 + nl), hoisted out of the 128-tile loop.
  // 16-bit A 16x32 layout (contiguous, matches B staging): lo lanes K {0..15}, hi lanes {16..31}.
  Frag16 afrag[8];
  {
    const int m = wave * 16 + nl;
    #pragma unroll
    for (int kb = 0; kb < 8; ++kb) {
      #pragma unroll
      for (int v = 0; v < 8; ++v) {
        int k0 = kb * 32 + 2 * v + (laneHi ? 16 : 0);
        afrag[kb].u[v] = *(const uint32_t*)&zsh[m * ZS + k0];
      }
    }
  }

  float minv[8];
  int   mini[8];
  #pragma unroll
  for (int r = 0; r < 8; ++r) { minv[r] = 3.0e38f; mini[r] = 0; }

  const int fragBase = lane * 8 + ((lane >> 3) & 1) * 4;

  for (int tile = 0; tile < 128; ++tile) {
    __syncthreads();   // (A) readers of slot (tile+2)%3 have finished

    // Issue tile+2 (or a harmless dummy re-read of tile 0 at the tail).
    {
      const int nt = (tile + 2 < 128) ? (tile + 2) : 0;
      const int ns = (tile + 2) % 3;
      const uint32_t* g = gthread + (size_t)nt * 2048;
      #pragma unroll
      for (int p = 0; p < 4; ++p)
        async_b128(g + p * 512, &csh[ns][ldw[p]]);
    }

    wait_asynccnt8();  // own chunks of `tile` complete (<= 2 newer tiles outstanding)
    __syncthreads();   // (B) everyone's chunks of `tile` complete

    // 16x16x256 tile: chain of 8 WMMA, B fragment = 2x ds_load_b128.
    const uint32_t* slotPtr = &csh[tile % 3][0];
    v8f acc = {};
    #pragma unroll
    for (int kb = 0; kb < 8; ++kb) {
      const uint32_t* p = slotPtr + kb * 264 + fragBase;
      uint4 lo = *(const uint4*)p;
      uint4 hi = *(const uint4*)(p + 4);
      Frag16 bfrag;
      bfrag.u[0] = lo.x; bfrag.u[1] = lo.y; bfrag.u[2] = lo.z; bfrag.u[3] = lo.w;
      bfrag.u[4] = hi.x; bfrag.u[5] = hi.y; bfrag.u[6] = hi.z; bfrag.u[7] = hi.w;
      acc = __builtin_amdgcn_wmma_f32_16x16x32_f16(
          false, afrag[kb].h, false, bfrag.h, (short)0, acc, false, false);
    }

    // score = ||c||^2 - 2 z.c ; strict < keeps lowest index on ties (n ascending).
    const int   nglob = tile * 16 + nl;
    const float cn    = cnorm[nglob];
    #pragma unroll
    for (int r = 0; r < 8; ++r) {
      float s = cn - 2.0f * acc[r];
      if (s < minv[r]) { minv[r] = s; mini[r] = nglob; }
    }
  }

  // Cross-lane argmin within each 16-lane half (C layout: lane = N, VGPR r = M row).
  #pragma unroll
  for (int r = 0; r < 8; ++r) {
    float v  = minv[r];
    int   id = mini[r];
    #pragma unroll
    for (int off = 1; off < 16; off <<= 1) {
      float ov  = __shfl_xor(v, off, 32);
      int   oid = __shfl_xor(id, off, 32);
      if (ov < v || (ov == v && oid < id)) { v = ov; id = oid; }
    }
    if (nl == 0) {
      int row = row0 + wave * 16 + (laneHi ? 8 : 0) + r;
      idx_out[row] = id;
    }
  }
}

// ------------------------------------------------------------------
// Kernel 2: gather codebook rows (exact f32), emit q = z + (c - z)
// in (b,d,h,w) layout, accumulate squared error (deterministic).
// ------------------------------------------------------------------
__global__ void __launch_bounds__(256) vq_gather(const float* __restrict__ z,
                                                 const float* __restrict__ cb,
                                                 const int* __restrict__ idx_in,
                                                 float* __restrict__ qout,
                                                 float* __restrict__ partials) {
  __shared__ float zq[32 * QS];
  __shared__ float wsum[8];

  const int t    = threadIdx.x;
  const int lane = t & 31;
  const int wave = t >> 5;
  const int row0 = blockIdx.x * 32;
  const int b    = row0 >> 10;
  const int hw0  = row0 & 1023;

  // Phase A: gather c[idx[row]] rows into LDS (coalesced codebook reads).
  #pragma unroll
  for (int it = 0; it < 4; ++it) {
    int rl = wave + 8 * it;
    int id = idx_in[row0 + rl];
    const float4* cp = (const float4*)(cb + (size_t)id * D_DIM);
    float4 u = cp[lane];
    float4 v = cp[lane + 32];
    float* d0 = &zq[rl * QS + lane * 4];
    d0[0] = u.x; d0[1] = u.y; d0[2] = u.z; d0[3] = u.w;
    float* d1 = &zq[rl * QS + 128 + lane * 4];
    d1[0] = v.x; d1[1] = v.y; d1[2] = v.z; d1[3] = v.w;
  }
  __syncthreads();

  // Phase B: whole wave shares d, lanes sweep hw -> coalesced z read / q write.
  const int hw = t & 31;
  const int dl = t >> 5;
  const float* zgb = z    + (size_t)b * (D_DIM * 1024) + hw0 + hw;
  float*       qgb = qout + (size_t)b * (D_DIM * 1024) + hw0 + hw;
  float acc = 0.0f;
  #pragma unroll
  for (int k = 0; k < 32; ++k) {
    int d = dl * 32 + k;
    float zv = zgb[(size_t)d * 1024];
    float cv = zq[hw * QS + d];
    float q  = zv + (cv - zv);   // reference straight-through rounding
    qgb[(size_t)d * 1024] = q;
    float df = zv - cv;
    acc += df * df;
  }

  #pragma unroll
  for (int off = 1; off < 32; off <<= 1) acc += __shfl_xor(acc, off, 32);
  if (lane == 0) wsum[wave] = acc;
  __syncthreads();
  if (t == 0) {
    float s = 0.0f;
    #pragma unroll
    for (int w = 0; w < 8; ++w) s += wsum[w];
    partials[blockIdx.x] = s;
  }
}

// ------------------------------------------------------------------
// Kernel 3: deterministic final reduction -> [loss, commitment_loss].
// ------------------------------------------------------------------
__global__ void vq_finalize(const float* __restrict__ partials,
                            float* __restrict__ out_losses) {
  float s = 0.0f;
  for (int i = 0; i < 512; ++i) s += partials[i];
  float mse = s * (1.0f / ((float)N_ROWS * (float)D_DIM));
  out_losses[0] = 0.25f * mse;  // BETA * commitment
  out_losses[1] = mse;          // commitment_loss
}

extern "C" void kernel_launch(void* const* d_in, const int* in_sizes, int n_in,
                              void* d_out, int out_size, void* d_ws, size_t ws_size,
                              hipStream_t stream) {
  const float* z  = (const float*)d_in[0];   // (16,256,32,32) f32
  const float* cb = (const float*)d_in[1];   // (2048,256) f32
  float* out = (float*)d_out;                // q (4,194,304) + loss + commitment

  char* ws = (char*)d_ws;
  uint32_t* cb16  = (uint32_t*)(ws);                               // 1,048,576 B
  float*    cnorm = (float*)(ws + 1048576);                        //     8,192 B
  int*      idx   = (int*)(ws + 1048576 + 8192);                   //    65,536 B
  float*    part  = (float*)(ws + 1048576 + 8192 + 65536);         //     2,048 B

  vq_prep    <<<K_CODES / 8, 256, 0, stream>>>(cb, cb16, cnorm);
  vq_argmin  <<<N_ROWS / 64, 128, 0, stream>>>(z, cb16, cnorm, idx);
  vq_gather  <<<N_ROWS / 32, 256, 0, stream>>>(z, cb, idx, out, part);
  vq_finalize<<<1, 1, 0, stream>>>(part, out + (size_t)N_ROWS * D_DIM);
}
